// FireflyVQ_70222715289747
// MI455X (gfx1250) — matrix operand order repaired and checked by reading, not compiled, not run on hardware
//
#include <hip/hip_runtime.h>

typedef __attribute__((ext_vector_type(16))) _Float16 v16h;
typedef __attribute__((ext_vector_type(8)))  _Float16 v8h;
typedef __attribute__((ext_vector_type(2)))  _Float16 h2;
typedef __attribute__((ext_vector_type(8)))  float    v8f;

static constexpr int kB   = 8;
static constexpr int kC   = 512;
static constexpr int kT   = 8192;
static constexpr int kT1  = 4096;   // after down layer 1
static constexpr int kT2  = 2048;   // after down layer 2
static constexpr int kNQ  = 9;
static constexpr int kCBS = 1024;
static constexpr int kCBD = 8;

static constexpr size_t kXOut   = (size_t)kB * kC * kT;        // 33,554,432 floats
static constexpr size_t kCodesN = (size_t)kB * kNQ * kT2;      // 147,456 ints
static constexpr size_t kN1     = (size_t)kB * kT1;            // 32768 (down L1 cols)
static constexpr size_t kN2     = (size_t)kB * kT2;            // 16384

// ---------------- workspace layout (bytes) ----------------
static constexpr size_t OFF_ACT  = 0;                               // f32 [B,512,4096]  67.1 MB
static constexpr size_t OFF_X0   = OFF_ACT + (size_t)kB*kC*kT1*4;   // f32 [B,512,2048]  33.6 MB
static constexpr size_t OFF_RES  = OFF_X0  + (size_t)kB*kC*kT2*4;   // f32 [B,512,2048]  33.6 MB
static constexpr size_t OFF_PX   = OFF_RES + (size_t)kB*kC*kT2*4;   // f16 pack, max 32768x1024 = 67.1 MB
static constexpr size_t OFF_PW   = OFF_PX  + kN1*1024*2;            // f16 weights, 1 MB
static constexpr size_t OFF_WIN  = OFF_PW  + (size_t)1024*512*2;    // f32 [9,8,512]
static constexpr size_t OFF_WOUT = OFF_WIN + (size_t)kNQ*kCBD*kC*4; // f32 [9,512,8]
static constexpr size_t OFF_CBN  = OFF_WOUT+ (size_t)kNQ*kC*kCBD*4; // f32 [9,1024,8]
static constexpr size_t OFF_LOSS = OFF_CBN + (size_t)kNQ*kCBS*kCBD*4;

// ---------------- small init / weight-norm kernels ----------------
__global__ void zero_loss_kernel(float* loss) { if (threadIdx.x == 0) loss[0] = 0.f; }

// weight_norm over rows: out[r,c] = g[r] * v[r,c] / ||v[r,:]||
__global__ void wn_kernel(const float* __restrict__ v, const float* __restrict__ g,
                          float* __restrict__ out, int cols) {
  __shared__ float red[128];
  int r = blockIdx.x;
  const float* vr = v + (size_t)r * cols;
  float s = 0.f;
  for (int c = threadIdx.x; c < cols; c += 128) { float x = vr[c]; s += x * x; }
  red[threadIdx.x] = s; __syncthreads();
  for (int off = 64; off > 0; off >>= 1) {
    if ((int)threadIdx.x < off) red[threadIdx.x] += red[threadIdx.x + off];
    __syncthreads();
  }
  float scale = g[r] / sqrtf(red[0]);
  for (int c = threadIdx.x; c < cols; c += 128) out[(size_t)r * cols + c] = vr[c] * scale;
}

// L2-normalize codebook rows (8 wide) with eps clamp
__global__ void cbn_kernel(const float* __restrict__ cb, float* __restrict__ out, int rows) {
  int r = blockIdx.x * blockDim.x + threadIdx.x;
  if (r >= rows) return;
  const float* p = cb + (size_t)r * kCBD;
  float s = 0.f;
#pragma unroll
  for (int d = 0; d < kCBD; ++d) s += p[d] * p[d];
  float inv = 1.f / fmaxf(sqrtf(s), 1e-12f);
#pragma unroll
  for (int d = 0; d < kCBD; ++d) out[(size_t)r * kCBD + d] = p[d] * inv;
}

// ---------------- packing kernels ----------------
__global__ void cvt_f32_f16_kernel(const float* __restrict__ in, _Float16* __restrict__ out, int n) {
  int i = blockIdx.x * blockDim.x + threadIdx.x;
  if (i < n) out[i] = (_Float16)in[i];
}

// up weights: out[(o*2+k)*512 + i] = w[i*1024 + o*2 + k]   (w = up_w[l], layout [i,o,k])
__global__ void pack_w_up_kernel(const float* __restrict__ w, _Float16* __restrict__ out) {
  int idx = blockIdx.x * blockDim.x + threadIdx.x;   // [0, 1024*512)
  int r = idx >> 9, i = idx & 511;
  int o = r >> 1, k = r & 1;
  out[idx] = (_Float16)w[(size_t)i * 1024 + o * 2 + k];
}

// down activations via LDS transpose: XT[(b*Th+t)*1024 + i*2+k] = x[(b*512+i)*Tin + 2t+k]
// tile: 32 i-rows x 64 t'-cols; reads and writes both >=128B contiguous per 32 lanes
__global__ __launch_bounds__(256)
void pack_x_dn_kernel(const float* __restrict__ x, _Float16* __restrict__ out, int Tin) {
  __shared__ unsigned tile[32][65];   // h2 packed as u32; pad 65 -> conflict-free columns
  int Th = Tin >> 1;
  int nt = Th >> 6;
  int bid = blockIdx.x;
  int tt = bid % nt;
  int it = (bid / nt) & 15;
  int b  = bid / (nt * 16);
  int i0 = it * 32, t0 = tt * 64;
  int tid = threadIdx.x;
  int tl = tid & 63, ib = tid >> 6;   // ib in 0..3
#pragma unroll
  for (int j = 0; j < 8; ++j) {
    int il = ib + j * 4;
    const float* p = x + ((size_t)(b * 512 + i0 + il)) * Tin + 2 * (t0 + tl);
    float2 v = *(const float2*)p;
    h2 o; o[0] = (_Float16)v.x; o[1] = (_Float16)v.y;
    union { h2 h; unsigned u; } cv; cv.h = o;
    tile[il][tl] = cv.u;
  }
  __syncthreads();
  int il2 = tid & 31, tb = tid >> 5;  // tb in 0..7
  h2* outp = (h2*)out;
#pragma unroll
  for (int j = 0; j < 8; ++j) {
    int tl2 = tb + j * 8;
    union { h2 h; unsigned u; } cv; cv.u = tile[il2][tl2];
    outp[((size_t)(b * Th + t0 + tl2)) * 512 + i0 + il2] = cv.h;
  }
}

// up activations via LDS transpose: XT[(b*Tin+t)*512 + i] = x[(b*512+i)*Tin + t] (- xsub)
// tile: 64 i-rows x 32 t-cols
__global__ __launch_bounds__(256)
void pack_x_up_kernel(const float* __restrict__ x, const float* __restrict__ xsub,
                      _Float16* __restrict__ out, int Tin) {
  __shared__ float tile[64][33];
  int nt = Tin >> 5;
  int bid = blockIdx.x;
  int tt = bid % nt;
  int it = (bid / nt) & 7;
  int b  = bid / (nt * 8);
  int i0 = it * 64, t0 = tt * 32;
  int tid = threadIdx.x;
  int tl = tid & 31, ib = tid >> 5;   // ib in 0..7
#pragma unroll
  for (int j = 0; j < 8; ++j) {
    int il = ib + j * 8;
    size_t s = ((size_t)(b * 512 + i0 + il)) * Tin + t0 + tl;
    float v = x[s];
    if (xsub) v -= xsub[s];
    tile[il][tl] = v;
  }
  __syncthreads();
  int ip = tid & 31, tb = tid >> 5;
  h2* outp = (h2*)out;
#pragma unroll
  for (int j = 0; j < 4; ++j) {
    int tl2 = tb + j * 8;
    h2 o; o[0] = (_Float16)tile[ip * 2][tl2]; o[1] = (_Float16)tile[ip * 2 + 1][tl2];
    outp[((size_t)(b * Tin + t0 + tl2)) * 256 + (i0 >> 1) + ip] = o;
  }
}

// ---------------- WMMA GEMM:  D[M,N] = A[M,K] * BT[N,K]^T + bias ----------------
// Register-blocked: each wave computes a 64(M) x 32(N) block = 4x2 WMMA tiles.
// mode 0 (down conv): out[(b*512 + m)*Tn + t],                 m in [0,512)
// mode 1 (up convT):  out[(b*512 + (m>>1))*(2Tn) + 2t+(m&1)],  m in [0,1024)
__device__ __forceinline__ v8f wmma16(v16h a, v16h b, v8f c) {
  return __builtin_amdgcn_wmma_f32_16x16x32_f16(false, a, false, b, (short)0, c, false, false);
}

__global__ __launch_bounds__(256)
void gemm_wmma_kernel(const _Float16* __restrict__ A, const _Float16* __restrict__ BT,
                      const float* __restrict__ bias, float* __restrict__ out,
                      float* __restrict__ out2, int Mblk, int Nblk, int K, int Tn, int mode) {
  int wave = (blockIdx.x * blockDim.x + threadIdx.x) >> 5;
  if (wave >= Mblk * Nblk) return;          // wave-uniform: EXEC stays all-ones for WMMA
  int lane = threadIdx.x & 31;
  int mb = wave % Mblk, nb = wave / Mblk;
  int ml = lane & 15, hi = lane >> 4;

  // A: ISA 16-bit A layout — lanes 0-15 hold K[0:8)+K[16:24), lanes 16-31 K[8:16)+K[24:32)
  const _Float16* pa0 = A + (size_t)(mb * 64 + ml) * K + hi * 8;
  const _Float16* pa1 = pa0 + (size_t)16 * K;
  const _Float16* pa2 = pa0 + (size_t)32 * K;
  const _Float16* pa3 = pa0 + (size_t)48 * K;
  // B: same K layout as A — lanes 0-15 hold K[0:8)+K[16:24), lanes 16-31 K[8:16)+K[24:32)
  const _Float16* pb0 = BT + (size_t)(nb * 32 + ml) * K + hi * 8;
  const _Float16* pb1 = pb0 + (size_t)16 * K;

  v8f acc[8] = {};
  for (int k = 0; k < K; k += 32) {
    v16h b0 = __builtin_shufflevector(*(const v8h*)(pb0 + k), *(const v8h*)(pb0 + k + 16), 0,1,2,3,4,5,6,7,8,9,10,11,12,13,14,15);
    v16h b1 = __builtin_shufflevector(*(const v8h*)(pb1 + k), *(const v8h*)(pb1 + k + 16), 0,1,2,3,4,5,6,7,8,9,10,11,12,13,14,15);
#define LOAD_A(p) __builtin_shufflevector(*(const v8h*)((p) + k), *(const v8h*)((p) + k + 16), \
                                          0,1,2,3,4,5,6,7,8,9,10,11,12,13,14,15)
    v16h a0 = LOAD_A(pa0);
    v16h a1 = LOAD_A(pa1);
    v16h a2 = LOAD_A(pa2);
    v16h a3 = LOAD_A(pa3);
#undef LOAD_A
    acc[0] = wmma16(a0, b0, acc[0]);
    acc[1] = wmma16(a0, b1, acc[1]);
    acc[2] = wmma16(a1, b0, acc[2]);
    acc[3] = wmma16(a1, b1, acc[3]);
    acc[4] = wmma16(a2, b0, acc[4]);
    acc[5] = wmma16(a2, b1, acc[5]);
    acc[6] = wmma16(a3, b0, acc[6]);
    acc[7] = wmma16(a3, b1, acc[7]);
  }

#pragma unroll
  for (int s = 0; s < 4; ++s) {
#pragma unroll
    for (int j = 0; j < 2; ++j) {
      int n  = nb * 32 + j * 16 + ml;
      int bb = n / Tn, t = n % Tn;
      int mbase = mb * 64 + s * 16 + (hi ? 8 : 0);
      v8f a = acc[s * 2 + j];
#pragma unroll
      for (int r = 0; r < 8; ++r) {
        int m = mbase + r;
        int o = (mode == 0) ? m : (m >> 1);
        float val = a[r] + bias[o];
        size_t addr = (mode == 0)
            ? ((size_t)(bb * 512 + o)) * Tn + t
            : ((size_t)(bb * 512 + o)) * (2 * (size_t)Tn) + 2 * t + (m & 1);
        out[addr] = val;
        if (out2) out2[addr] = val;
      }
    }
  }
}

// ---------------- VQ stage kernels ----------------
// fused in_proj (1x1 weight-normed conv) + nearest-code search + loss + code emit
__global__ __launch_bounds__(256)
void vq_encode_kernel(const float* __restrict__ res, const float* __restrict__ w_in,
                      const float* __restrict__ in_b, const float* __restrict__ cbn,
                      const float* __restrict__ cbraw, int* __restrict__ codes,
                      float* __restrict__ loss_accum, int stage) {
  __shared__ float sw[kCBD * kC];       // 16 KB
  __shared__ float scb[kCBS * kCBD];    // 32 KB (normalized codebook)
  __shared__ float sbias[kCBD];
  __shared__ float red[256];
  int tid = threadIdx.x;
  for (int i = tid; i < kCBD * kC; i += 256) sw[i] = w_in[i];
  for (int i = tid; i < kCBS * kCBD; i += 256) scb[i] = cbn[i];
  if (tid < kCBD) sbias[tid] = in_b[tid];
  __syncthreads();

  int n = blockIdx.x * 256 + tid;       // n in [0, B*T2)
  int b = n >> 11, t = n & (kT2 - 1);
  const float* rp = res + ((size_t)b * kC) * kT2 + t;

  float z[kCBD];
#pragma unroll
  for (int d = 0; d < kCBD; ++d) z[d] = sbias[d];
  for (int c = 0; c < kC; ++c) {
    float v = rp[(size_t)c * kT2];
#pragma unroll
    for (int d = 0; d < kCBD; ++d) z[d] += v * sw[d * kC + c];
  }
  float ss = 0.f;
#pragma unroll
  for (int d = 0; d < kCBD; ++d) ss += z[d] * z[d];
  float inv = 1.f / fmaxf(sqrtf(ss), 1e-12f);
  float e[kCBD];
#pragma unroll
  for (int d = 0; d < kCBD; ++d) e[d] = z[d] * inv;

  float best = -3.0e38f; int bi = 0;
  for (int j = 0; j < kCBS; ++j) {
    const float* cj = scb + j * kCBD;
    float dot = 0.f;
#pragma unroll
    for (int d = 0; d < kCBD; ++d) dot += e[d] * cj[d];
    if (dot > best) { best = dot; bi = j; }   // strict: first-max tie-break
  }

  const float* q = cbraw + (size_t)bi * kCBD;
  float l = 0.f;
#pragma unroll
  for (int d = 0; d < kCBD; ++d) { float df = z[d] - q[d]; l += df * df; }
  codes[((size_t)b * kNQ + stage) * kT2 + t] = bi;

  red[tid] = l; __syncthreads();
  for (int off = 128; off > 0; off >>= 1) {
    if (tid < off) red[tid] += red[tid + off];
    __syncthreads();
  }
  if (tid == 0) atomicAdd(loss_accum, red[0]);
}

// out_proj (1x1 weight-normed conv of gathered codebook rows) folded into residual update
__global__ __launch_bounds__(256)
void vq_update_kernel(float* __restrict__ res, const float* __restrict__ w_out,
                      const float* __restrict__ out_b, const float* __restrict__ cbraw,
                      const int* __restrict__ codes, int stage) {
  __shared__ float sw[kC * kCBD];       // 16 KB
  __shared__ float sb[kC];
  int tid = threadIdx.x;
  for (int i = tid; i < kC * kCBD; i += 256) sw[i] = w_out[i];
  for (int i = tid; i < kC; i += 256) sb[i] = out_b[i];
  __syncthreads();

  int n = blockIdx.x * 256 + tid;
  int b = n >> 11, t = n & (kT2 - 1);
  int idx = codes[((size_t)b * kNQ + stage) * kT2 + t];
  float q[kCBD];
  const float* qp = cbraw + (size_t)idx * kCBD;
#pragma unroll
  for (int d = 0; d < kCBD; ++d) q[d] = qp[d];

  float* rp = res + ((size_t)b * kC) * kT2 + t;
  for (int c = 0; c < kC; ++c) {
    float v = sb[c];
#pragma unroll
    for (int d = 0; d < kCBD; ++d) v += q[d] * sw[c * kCBD + d];
    rp[(size_t)c * kT2] -= v;           // residual -= z_q_out  (z_q kept implicitly)
  }
}

__global__ void finalize_loss_kernel(const float* __restrict__ loss_accum, float* __restrict__ out) {
  if (threadIdx.x == 0) {
    float m = loss_accum[0] * (1.0f / (float)(kB * kCBD * kT2));
    out[0] = m;   // commit_loss
    out[1] = m;   // cb_loss (identical in forward values)
  }
}

// ---------------- host orchestration ----------------
extern "C" void kernel_launch(void* const* d_in, const int* in_sizes, int n_in,
                              void* d_out, int out_size, void* d_ws, size_t ws_size,
                              hipStream_t stream) {
  const float* z      = (const float*)d_in[0];
  const float* down_w = (const float*)d_in[1];
  const float* down_b = (const float*)d_in[2];
  const float* up_w   = (const float*)d_in[3];
  const float* up_b   = (const float*)d_in[4];
  const float* in_v   = (const float*)d_in[5];
  const float* in_g   = (const float*)d_in[6];
  const float* in_b   = (const float*)d_in[7];
  const float* out_v  = (const float*)d_in[8];
  const float* out_g  = (const float*)d_in[9];
  const float* out_b  = (const float*)d_in[10];
  const float* cb     = (const float*)d_in[11];

  char* ws = (char*)d_ws;
  float*    act   = (float*)(ws + OFF_ACT);
  float*    x0    = (float*)(ws + OFF_X0);
  float*    res   = (float*)(ws + OFF_RES);
  _Float16* packX = (_Float16*)(ws + OFF_PX);
  _Float16* packW = (_Float16*)(ws + OFF_PW);
  float*    w_in  = (float*)(ws + OFF_WIN);
  float*    w_out = (float*)(ws + OFF_WOUT);
  float*    cbn   = (float*)(ws + OFF_CBN);
  float*    loss  = (float*)(ws + OFF_LOSS);

  float* xout  = (float*)d_out;
  int*   codes = (int*)(xout + kXOut);
  float* louts = xout + kXOut + kCodesN;

  // --- precompute: weight-norm projections, normalized codebooks, loss = 0 ---
  zero_loss_kernel<<<1, 32, 0, stream>>>(loss);
  wn_kernel<<<kNQ * kCBD, 128, 0, stream>>>(in_v, in_g, w_in, kC);     // 72 rows x 512
  wn_kernel<<<kNQ * kC, 128, 0, stream>>>(out_v, out_g, w_out, kCBD);  // 4608 rows x 8
  cbn_kernel<<<(kNQ * kCBS + 255) / 256, 256, 0, stream>>>(cb, cbn, kNQ * kCBS);

  // --- downsample layer 1: M=512, K=1024, N = B*4096 ---
  cvt_f32_f16_kernel<<<(512 * 1024 + 255) / 256, 256, 0, stream>>>(down_w, packW, 512 * 1024);
  {
    int grid = kB * 16 * ((kT / 2) / 64);                 // 8192 tiles
    pack_x_dn_kernel<<<grid, 256, 0, stream>>>(z, packX, kT);
    int waves = (512 / 64) * (int)(kN1 / 32);             // 8 * 1024
    gemm_wmma_kernel<<<waves / 8, 256, 0, stream>>>(packW, packX, down_b, act, nullptr,
                                                    512 / 64, (int)(kN1 / 32), 1024, kT1, 0);
  }
  // --- downsample layer 2: N = B*2048; dual-write x0 and residual ---
  cvt_f32_f16_kernel<<<(512 * 1024 + 255) / 256, 256, 0, stream>>>(down_w + 512 * 1024, packW, 512 * 1024);
  {
    int grid = kB * 16 * ((kT1 / 2) / 64);                // 4096 tiles
    pack_x_dn_kernel<<<grid, 256, 0, stream>>>(act, packX, kT1);
    int waves = (512 / 64) * (int)(kN2 / 32);             // 8 * 512
    gemm_wmma_kernel<<<waves / 8, 256, 0, stream>>>(packW, packX, down_b + 512, x0, res,
                                                    512 / 64, (int)(kN2 / 32), 1024, kT2, 0);
  }

  // --- 9 residual VQ stages ---
  for (int i = 0; i < kNQ; ++i) {
    vq_encode_kernel<<<(int)(kN2 / 256), 256, 0, stream>>>(
        res, w_in + (size_t)i * kCBD * kC, in_b + (size_t)i * kCBD,
        cbn + (size_t)i * kCBS * kCBD, cb + (size_t)i * kCBS * kCBD, codes, loss, i);
    vq_update_kernel<<<(int)(kN2 / 256), 256, 0, stream>>>(
        res, w_out + (size_t)i * kC * kCBD, out_b + (size_t)i * kC,
        cb + (size_t)i * kCBS * kCBD, codes, i);
  }

  // --- upsample layer 1: input z_q = x0 - res; M=1024, K=512, N=B*2048 -> act[B,512,4096] ---
  pack_w_up_kernel<<<(1024 * 512) / 256, 256, 0, stream>>>(up_w, packW);
  {
    int grid = kB * 8 * (kT2 / 32);                       // 4096 tiles
    pack_x_up_kernel<<<grid, 256, 0, stream>>>(x0, res, packX, kT2);
    int waves = (1024 / 64) * (int)(kN2 / 32);            // 16 * 512
    gemm_wmma_kernel<<<waves / 8, 256, 0, stream>>>(packW, packX, up_b, act, nullptr,
                                                    1024 / 64, (int)(kN2 / 32), 512, kT2, 1);
  }
  // --- upsample layer 2: M=1024, K=512, N=B*4096 -> d_out [B,512,8192] ---
  pack_w_up_kernel<<<(1024 * 512) / 256, 256, 0, stream>>>(up_w + 512 * 1024, packW);
  {
    int grid = kB * 8 * (kT1 / 32);                       // 8192 tiles
    pack_x_up_kernel<<<grid, 256, 0, stream>>>(act, nullptr, packX, kT1);
    int waves = (1024 / 64) * (int)(kN1 / 32);            // 16 * 1024
    gemm_wmma_kernel<<<waves / 8, 256, 0, stream>>>(packW, packX, up_b + 512, xout, nullptr,
                                                    1024 / 64, (int)(kN1 / 32), 512, kT1, 1);
  }

  finalize_loss_kernel<<<1, 32, 0, stream>>>(loss, louts);
}
